// GAT_13718125543872
// MI455X (gfx1250) — compile-verified, matrix operand order checked by reading, not run on hardware
//
#include <hip/hip_runtime.h>
#include <hip/hip_bf16.h>
#include <stdint.h>

#define N_NODES 50000
#define N_EDGES 800000
#define ETOT    (N_EDGES + N_NODES)
#define F_IN    256
#define HEADS   8
#define HID     16
#define D1      (HEADS * HID)   // 128
#define F_OUT   64
#define NEG_SLOPE 0.2f
#define LN_EPS  1e-5f
#define SM_EPS  1e-16f

typedef __attribute__((ext_vector_type(16))) __bf16 v16bf;
typedef __attribute__((ext_vector_type(8)))  float  v8f;

// ---------- helpers ----------
__device__ __forceinline__ unsigned short f2bf(float f) {
  unsigned u = __float_as_uint(f);
  u += 0x7FFFu + ((u >> 16) & 1u);          // round-to-nearest-even
  return (unsigned short)(u >> 16);
}
__device__ __forceinline__ float warp_sum(float v) {
#pragma unroll
  for (int m = 16; m >= 1; m >>= 1) v += __shfl_xor(v, m, 32);
  return v;
}
__device__ __forceinline__ float warp_max(float v) {
#pragma unroll
  for (int m = 16; m >= 1; m >>= 1) v = fmaxf(v, __shfl_xor(v, m, 32));
  return v;
}
// monotonic float <-> ordered-uint key (for atomicMax on floats incl. negatives)
__device__ __forceinline__ unsigned fkey(float f) {
  unsigned u = __float_as_uint(f);
  return (u & 0x80000000u) ? ~u : (u | 0x80000000u);
}
__device__ __forceinline__ float fdec(unsigned k) {
  unsigned u = (k & 0x80000000u) ? (k & 0x7FFFFFFFu) : ~k;
  return __uint_as_float(u);
}
__device__ __forceinline__ void edge_sd(const long long* __restrict__ ei, int e,
                                        int* s, int* d) {
  if (e < N_EDGES) { *s = (int)ei[e]; *d = (int)ei[N_EDGES + e]; }
  else             { *s = e - N_EDGES; *d = e - N_EDGES; }   // self-loops
}
// 16-bit WMMA A/B fragment K-offset for dword pair t (0..7), lane half (0/1)
__device__ __forceinline__ int kofs(int t, int half) {
  return ((t & 4) ? 16 : 0) + half * 8 + (t & 3) * 2;
}

// ---------- K1: LayerNorm(256) -> bf16 ----------
__global__ void k_ln1(const float* __restrict__ x, const float* __restrict__ w,
                      const float* __restrict__ b, unsigned short* __restrict__ xnb) {
  int wave = threadIdx.x >> 5, lane = threadIdx.x & 31;
  int n = blockIdx.x * 8 + wave;
  if (n >= N_NODES) return;
  const float* row = x + (size_t)n * F_IN;
  float v[8]; float s = 0.f;
#pragma unroll
  for (int t = 0; t < 8; ++t) { v[t] = row[lane + t * 32]; s += v[t]; }
  float mu = warp_sum(s) * (1.f / F_IN);
  float q = 0.f;
#pragma unroll
  for (int t = 0; t < 8; ++t) { float d = v[t] - mu; q += d * d; }
  float rs = rsqrtf(warp_sum(q) * (1.f / F_IN) + LN_EPS);
  unsigned short* orow = xnb + (size_t)n * F_IN;
#pragma unroll
  for (int t = 0; t < 8; ++t) {
    int c = lane + t * 32;
    orow[c] = f2bf((v[t] - mu) * rs * w[c] + b[c]);
  }
}

// ---------- weight transpose+convert: W[K,Nc] f32 -> Wt[Nc,K] bf16 ----------
__global__ void k_convT(const float* __restrict__ W, unsigned short* __restrict__ Wt,
                        int K, int Nc) {
  int i = blockIdx.x * blockDim.x + threadIdx.x;
  if (i >= K * Nc) return;
  int k = i / Nc, n = i % Nc;
  Wt[(size_t)n * K + k] = f2bf(W[i]);
}

// ---------- WMMA GEMM: A[M,K]bf16 x Bt[Nc,K]bf16 -> C[M,Nc]f32 ----------
__global__ void k_gemm_wmma(const unsigned short* __restrict__ A,
                            const unsigned short* __restrict__ Bt,
                            float* __restrict__ C,
                            int M, int K, int Nc, int rowTilesPerBlock) {
  int colTiles = Nc >> 4;
  int wave = threadIdx.x >> 5, lane = threadIdx.x & 31;
  int rowTile = blockIdx.x * rowTilesPerBlock + wave / colTiles;
  int colTile = wave % colTiles;
  if (rowTile * 16 >= M) return;                 // wave-uniform: EXEC stays all-1s
  int half = lane >> 4, sub = lane & 15;
  const unsigned short* arow = A  + (size_t)(rowTile * 16 + sub) * K;
  const unsigned short* brow = Bt + (size_t)(colTile * 16 + sub) * K;
  v8f acc = {};
  for (int k0 = 0; k0 < K; k0 += 32) {
    union { v16bf v; unsigned u[8]; } a, b;
#pragma unroll
    for (int t = 0; t < 8; ++t) {
      a.u[t] = *(const unsigned*)(arow + k0 + kofs(t, half));
      b.u[t] = *(const unsigned*)(brow + k0 + kofs(t, half));
    }
    acc = __builtin_amdgcn_wmma_f32_16x16x32_bf16(
        false, a.v, false, b.v, (short)0, acc, false, false);
  }
  float* crow = C + (size_t)(rowTile * 16 + half * 8) * Nc + colTile * 16 + sub;
#pragma unroll
  for (int r = 0; r < 8; ++r) crow[(size_t)r * Nc] = acc[r];
}

// ---------- attention coefficients, layer 1 ----------
__global__ void k_att1(const float* __restrict__ h1, const float* __restrict__ att_s,
                       const float* __restrict__ att_d,
                       float* __restrict__ as1, float* __restrict__ ad1) {
  int i = blockIdx.x * blockDim.x + threadIdx.x;       // n*HEADS + h
  if (i >= N_NODES * HEADS) return;
  int n = i >> 3, h = i & 7;
  const float* hp = h1 + (size_t)n * D1 + h * HID;
  const float* sp = att_s + h * HID;
  const float* dp = att_d + h * HID;
  float s = 0.f, d = 0.f;
#pragma unroll
  for (int c = 0; c < HID; ++c) { float v = hp[c]; s += v * sp[c]; d += v * dp[c]; }
  as1[i] = s; ad1[i] = d;
}

// ---------- edge pass A: segment max (layer 1) ----------
__global__ void k_emax1(const long long* __restrict__ ei, const float* __restrict__ as1,
                        const float* __restrict__ ad1, unsigned* __restrict__ mk) {
  int e = blockIdx.x * blockDim.x + threadIdx.x;
  if (e >= ETOT) return;
  int s, d; edge_sd(ei, e, &s, &d);
#pragma unroll
  for (int h = 0; h < HEADS; ++h) {
    float v = as1[s * HEADS + h] + ad1[d * HEADS + h];
    v = v > 0.f ? v : NEG_SLOPE * v;
    atomicMax(&mk[d * HEADS + h], fkey(v));
  }
}

// ---------- edge pass B: segment sum of exp (layer 1) ----------
__global__ void k_esum1(const long long* __restrict__ ei, const float* __restrict__ as1,
                        const float* __restrict__ ad1, const unsigned* __restrict__ mk,
                        float* __restrict__ ssum) {
  int e = blockIdx.x * blockDim.x + threadIdx.x;
  if (e >= ETOT) return;
  int s, d; edge_sd(ei, e, &s, &d);
#pragma unroll
  for (int h = 0; h < HEADS; ++h) {
    float v = as1[s * HEADS + h] + ad1[d * HEADS + h];
    v = v > 0.f ? v : NEG_SLOPE * v;
    atomicAdd(&ssum[d * HEADS + h], __expf(v - fdec(mk[d * HEADS + h])));
  }
}

// ---------- edge pass C: weighted aggregation (layer 1) ----------
__global__ void k_eagg1(const long long* __restrict__ ei, const float* __restrict__ as1,
                        const float* __restrict__ ad1, const unsigned* __restrict__ mk,
                        const float* __restrict__ ssum, const float* __restrict__ h1,
                        float* __restrict__ out1) {
  int i = blockIdx.x * blockDim.x + threadIdx.x;       // e*HEADS + h
  if (i >= ETOT * HEADS) return;
  int e = i >> 3, h = i & 7;
  int s, d; edge_sd(ei, e, &s, &d);
  float v = as1[s * HEADS + h] + ad1[d * HEADS + h];
  v = v > 0.f ? v : NEG_SLOPE * v;
  float alpha = __expf(v - fdec(mk[d * HEADS + h])) / (ssum[d * HEADS + h] + SM_EPS);
  const float4* hp = (const float4*)(h1 + (size_t)s * D1 + h * HID);
  float* op = out1 + (size_t)d * D1 + h * HID;
#pragma unroll
  for (int q = 0; q < 4; ++q) {
    float4 m = hp[q];
    atomicAdd(op + q * 4 + 0, m.x * alpha);
    atomicAdd(op + q * 4 + 1, m.y * alpha);
    atomicAdd(op + q * 4 + 2, m.z * alpha);
    atomicAdd(op + q * 4 + 3, m.w * alpha);
  }
}

// ---------- bias + ELU + LayerNorm(128) -> bf16 ----------
__global__ void k_post1(const float* __restrict__ out1, const float* __restrict__ b1,
                        const float* __restrict__ nw, const float* __restrict__ nb,
                        unsigned short* __restrict__ h1n) {
  int wave = threadIdx.x >> 5, lane = threadIdx.x & 31;
  int n = blockIdx.x * 8 + wave;
  if (n >= N_NODES) return;
  const float* row = out1 + (size_t)n * D1;
  float v[4]; float s = 0.f;
#pragma unroll
  for (int t = 0; t < 4; ++t) {
    int c = lane + t * 32;
    float u = row[c] + b1[c];
    u = u > 0.f ? u : (__expf(u) - 1.f);               // ELU(alpha=1)
    v[t] = u; s += u;
  }
  float mu = warp_sum(s) * (1.f / D1);
  float q = 0.f;
#pragma unroll
  for (int t = 0; t < 4; ++t) { float dd = v[t] - mu; q += dd * dd; }
  float rs = rsqrtf(warp_sum(q) * (1.f / D1) + LN_EPS);
  unsigned short* orow = h1n + (size_t)n * D1;
#pragma unroll
  for (int t = 0; t < 4; ++t) {
    int c = lane + t * 32;
    orow[c] = f2bf((v[t] - mu) * rs * nw[c] + nb[c]);
  }
}

// ---------- attention coefficients, layer 2 (1 head, 64 ch) ----------
__global__ void k_att2(const float* __restrict__ g2, const float* __restrict__ att_s,
                       const float* __restrict__ att_d,
                       float* __restrict__ as2, float* __restrict__ ad2) {
  int n = blockIdx.x * blockDim.x + threadIdx.x;
  if (n >= N_NODES) return;
  const float* gp = g2 + (size_t)n * F_OUT;
  float s = 0.f, d = 0.f;
#pragma unroll
  for (int c = 0; c < F_OUT; ++c) { float v = gp[c]; s += v * att_s[c]; d += v * att_d[c]; }
  as2[n] = s; ad2[n] = d;
}

__global__ void k_emax2(const long long* __restrict__ ei, const float* __restrict__ as2,
                        const float* __restrict__ ad2, unsigned* __restrict__ mk) {
  int e = blockIdx.x * blockDim.x + threadIdx.x;
  if (e >= ETOT) return;
  int s, d; edge_sd(ei, e, &s, &d);
  float v = as2[s] + ad2[d];
  v = v > 0.f ? v : NEG_SLOPE * v;
  atomicMax(&mk[d], fkey(v));
}

__global__ void k_esum2(const long long* __restrict__ ei, const float* __restrict__ as2,
                        const float* __restrict__ ad2, const unsigned* __restrict__ mk,
                        float* __restrict__ ssum) {
  int e = blockIdx.x * blockDim.x + threadIdx.x;
  if (e >= ETOT) return;
  int s, d; edge_sd(ei, e, &s, &d);
  float v = as2[s] + ad2[d];
  v = v > 0.f ? v : NEG_SLOPE * v;
  atomicAdd(&ssum[d], __expf(v - fdec(mk[d])));
}

__global__ void k_eagg2(const long long* __restrict__ ei, const float* __restrict__ as2,
                        const float* __restrict__ ad2, const unsigned* __restrict__ mk,
                        const float* __restrict__ ssum, const float* __restrict__ g2,
                        float* __restrict__ out2) {
  int e = blockIdx.x * blockDim.x + threadIdx.x;
  if (e >= ETOT) return;
  int s, d; edge_sd(ei, e, &s, &d);
  float v = as2[s] + ad2[d];
  v = v > 0.f ? v : NEG_SLOPE * v;
  float alpha = __expf(v - fdec(mk[d])) / (ssum[d] + SM_EPS);
  const float4* gp = (const float4*)(g2 + (size_t)s * F_OUT);
  float* op = out2 + (size_t)d * F_OUT;
#pragma unroll
  for (int q = 0; q < 16; ++q) {
    float4 m = gp[q];
    atomicAdd(op + q * 4 + 0, m.x * alpha);
    atomicAdd(op + q * 4 + 1, m.y * alpha);
    atomicAdd(op + q * 4 + 2, m.z * alpha);
    atomicAdd(op + q * 4 + 3, m.w * alpha);
  }
}

// ---------- bias + log_softmax(64) ----------
__global__ void k_final(const float* __restrict__ out2, const float* __restrict__ b2,
                        float* __restrict__ out) {
  int wave = threadIdx.x >> 5, lane = threadIdx.x & 31;
  int n = blockIdx.x * 8 + wave;
  if (n >= N_NODES) return;
  const float* row = out2 + (size_t)n * F_OUT;
  float v0 = row[lane]      + b2[lane];
  float v1 = row[lane + 32] + b2[lane + 32];
  float mx = warp_max(fmaxf(v0, v1));
  float se = warp_sum(__expf(v0 - mx) + __expf(v1 - mx));
  float lse = mx + __logf(se);
  float* orow = out + (size_t)n * F_OUT;
  orow[lane]      = v0 - lse;
  orow[lane + 32] = v1 - lse;
}

// ---------- host ----------
extern "C" void kernel_launch(void* const* d_in, const int* in_sizes, int n_in,
                              void* d_out, int out_size, void* d_ws, size_t ws_size,
                              hipStream_t stream) {
  (void)in_sizes; (void)n_in; (void)out_size; (void)ws_size;
  const float*     x    = (const float*)d_in[0];
  const long long* ei   = (const long long*)d_in[1];
  const float*     n1w  = (const float*)d_in[2];
  const float*     n1b  = (const float*)d_in[3];
  const float*     W1   = (const float*)d_in[4];
  const float*     as1w = (const float*)d_in[5];
  const float*     ad1w = (const float*)d_in[6];
  const float*     b1   = (const float*)d_in[7];
  const float*     n2w  = (const float*)d_in[8];
  const float*     n2b  = (const float*)d_in[9];
  const float*     W2   = (const float*)d_in[10];
  const float*     as2w = (const float*)d_in[11];
  const float*     ad2w = (const float*)d_in[12];
  const float*     b2   = (const float*)d_in[13];

  char* ws = (char*)d_ws;
  size_t off = 0;
  auto alloc = [&](size_t bytes) -> char* {
    char* p = ws + off; off += (bytes + 255) & ~(size_t)255; return p;
  };
  unsigned short* xnb = (unsigned short*)alloc((size_t)N_NODES * F_IN * 2);
  unsigned short* w1t = (unsigned short*)alloc((size_t)F_IN * D1 * 2);
  float*          h1  = (float*)alloc((size_t)N_NODES * D1 * 4);
  float*          as1 = (float*)alloc((size_t)N_NODES * HEADS * 4);
  float*          ad1 = (float*)alloc((size_t)N_NODES * HEADS * 4);
  unsigned short* h1n = (unsigned short*)alloc((size_t)N_NODES * D1 * 2);
  unsigned short* w2t = (unsigned short*)alloc((size_t)D1 * F_OUT * 2);
  float*          g2  = (float*)alloc((size_t)N_NODES * F_OUT * 4);
  float*          as2 = (float*)alloc((size_t)N_NODES * 4);
  float*          ad2 = (float*)alloc((size_t)N_NODES * 4);
  // zero-initialized accumulator region (contiguous -> single memset)
  char* zbase = ws + off;
  float*    out1 = (float*)alloc((size_t)N_NODES * D1 * 4);
  float*    s1   = (float*)alloc((size_t)N_NODES * HEADS * 4);
  unsigned* mk1  = (unsigned*)alloc((size_t)N_NODES * HEADS * 4);
  float*    out2 = (float*)alloc((size_t)N_NODES * F_OUT * 4);
  float*    s2   = (float*)alloc((size_t)N_NODES * 4);
  unsigned* mk2  = (unsigned*)alloc((size_t)N_NODES * 4);
  size_t zbytes = (size_t)((ws + off) - zbase);
  hipMemsetAsync(zbase, 0, zbytes, stream);

  const int B = 256;
  const int nodeWaveBlocks = (N_NODES + 7) / 8;        // 8 waves/block, 1 node/wave
  const int edgeBlocks = (ETOT + B - 1) / B;

  // layer 1
  k_ln1<<<nodeWaveBlocks, B, 0, stream>>>(x, n1w, n1b, xnb);
  k_convT<<<(F_IN * D1 + B - 1) / B, B, 0, stream>>>(W1, w1t, F_IN, D1);
  k_gemm_wmma<<<N_NODES / 16, B, 0, stream>>>(xnb, w1t, h1, N_NODES, F_IN, D1, 1);
  k_att1<<<(N_NODES * HEADS + B - 1) / B, B, 0, stream>>>(h1, as1w, ad1w, as1, ad1);
  k_emax1<<<edgeBlocks, B, 0, stream>>>(ei, as1, ad1, mk1);
  k_esum1<<<edgeBlocks, B, 0, stream>>>(ei, as1, ad1, mk1, s1);
  k_eagg1<<<(ETOT * HEADS + B - 1) / B, B, 0, stream>>>(ei, as1, ad1, mk1, s1, h1, out1);
  k_post1<<<nodeWaveBlocks, B, 0, stream>>>(out1, b1, n2w, n2b, h1n);

  // layer 2
  k_convT<<<(D1 * F_OUT + B - 1) / B, B, 0, stream>>>(W2, w2t, D1, F_OUT);
  k_gemm_wmma<<<(N_NODES / 16 + 1) / 2, B, 0, stream>>>(h1n, w2t, g2, N_NODES, D1, F_OUT, 2);
  k_att2<<<(N_NODES + B - 1) / B, B, 0, stream>>>(g2, as2w, ad2w, as2, ad2);
  k_emax2<<<edgeBlocks, B, 0, stream>>>(ei, as2, ad2, mk2);
  k_esum2<<<edgeBlocks, B, 0, stream>>>(ei, as2, ad2, mk2, s2);
  k_eagg2<<<edgeBlocks, B, 0, stream>>>(ei, as2, ad2, mk2, s2, g2, out2);
  k_final<<<nodeWaveBlocks, B, 0, stream>>>(out2, b2, (float*)d_out);
}
